// TriangMul_80711025426883
// MI455X (gfx1250) — matrix operand order repaired and checked by reading, not compiled, not run on hardware
//
#include <hip/hip_runtime.h>
#include <hip/hip_bf16.h>
#include <math.h>

typedef __attribute__((ext_vector_type(2)))  float  v2f;
typedef __attribute__((ext_vector_type(8)))  float  v8f;
typedef __attribute__((ext_vector_type(8)))  __bf16 v8bf;
typedef __attribute__((ext_vector_type(16))) __bf16 v16bf;
typedef __attribute__((ext_vector_type(4)))  int    v4i;
typedef __attribute__((address_space(1))) v4i* as1_v4i;   // global int4*
typedef __attribute__((address_space(3))) v4i* as3_v4i;   // LDS int4*

#define CCH    128
#define LDIM   512
#define LL     (LDIM * LDIM)         /* 262144 positions */
#define WPITCH 132                   /* LDS row pitch in floats: 4-word bank stride */
#define LN_EPS 1e-5f

__device__ __forceinline__ float sigmoidf(float v) {
    return 1.0f / (1.0f + __expf(-v));
}

// --- CDNA5 async global->LDS copy (16B per lane) + wait -------------------
__device__ __forceinline__ void async_copy_b128(const void* g, void* l) {
#if __has_builtin(__builtin_amdgcn_global_load_async_to_lds_b128)
    __builtin_amdgcn_global_load_async_to_lds_b128(
        (as1_v4i)(uintptr_t)g,
        (as3_v4i)(unsigned)(uintptr_t)l, 0, 0);
#else
    unsigned lo = (unsigned)(uintptr_t)l;          // LDS byte offset (low 32 bits)
    unsigned long long ga = (unsigned long long)(uintptr_t)g;
    asm volatile("global_load_async_to_lds_b128 %0, %1, off"
                 :: "v"(lo), "v"(ga) : "memory");
#endif
}

__device__ __forceinline__ void wait_async0() {
#if __has_builtin(__builtin_amdgcn_s_wait_asynccnt)
    __builtin_amdgcn_s_wait_asynccnt(0);
#else
    asm volatile("s_wait_asynccnt 0x0" ::: "memory");
#endif
}

// ---------------------------------------------------------------------------
// Kernel 1: fused gated projections.
//   left  = sigmoid(x@Wl1^T + bl1) * (x@Wl2^T + bl2)
//   right = sigmoid(x@Wr1^T + br1) * (x@Wr2^T + br2)
// The four 128x128 weight matrices are staged once per block into LDS
// (pitch-132 rows, async global->LDS copies), so the fp32-WMMA inner loop has
// one global A-load + four conflict-free ds_load_b64 B-fragments per k-step.
// Results stored bf16 in [c][row][k] layout for transpose-free kernel 2 loads.
// ---------------------------------------------------------------------------
__global__ __launch_bounds__(256) void proj_kernel(
    const float* __restrict__ x,
    const float* __restrict__ Wl1, const float* __restrict__ bl1,
    const float* __restrict__ Wl2, const float* __restrict__ bl2,
    const float* __restrict__ Wr1, const float* __restrict__ br1,
    const float* __restrict__ Wr2, const float* __restrict__ br2,
    __bf16* __restrict__ Lbf, __bf16* __restrict__ Rbf)
{
    extern __shared__ float smemw[];         // 4 * 128 * WPITCH floats
    const int lane  = threadIdx.x & 31;
    const int wave  = threadIdx.x >> 5;
    const int laneM = lane & 15;
    const int klo   = (lane < 16) ? 0 : 2;   // f32 WMMA: hi half-wave holds K=2,3
    const int mhi   = (lane < 16) ? 0 : 8;   // D tile: hi half-wave holds M=8..15

    // ---- stage the 4 weight matrices into LDS (64 rows per wave) ----
    {
        const int m = wave >> 1;                       // matrix id (uniform/wave)
        const float* wsrc = (m == 0) ? Wl1 : (m == 1) ? Wl2 : (m == 2) ? Wr1 : Wr2;
        const int r0 = (wave & 1) * 64;                // first row for this wave
        #pragma unroll 8
        for (int t = 0; t < 64; ++t) {
            const int r = r0 + t;
            async_copy_b128(wsrc + (size_t)r * CCH + lane * 4,
                            smemw + ((size_t)(m * CCH + r) * WPITCH + lane * 4));
        }
        wait_async0();
    }
    __syncthreads();

    const int tile = blockIdx.x * 8 + wave;
    const int p0   = tile * 16;
    const int a    = p0 >> 9;                // grid row
    const int b0   = p0 & 511;               // grid col base (multiple of 16)

    const float* xrow = x + (size_t)(p0 + laneM) * CCH;

    for (int nt = 0; nt < 8; ++nt) {
        const int c0 = nt * 16;
        v8f acc1 = {}, acc2 = {}, acc3 = {}, acc4 = {};
        // per-lane LDS rows: channel c0+laneM of each staged matrix
        const float* s1 = smemw + (size_t)(0 * CCH + c0 + laneM) * WPITCH + klo;
        const float* s2 = smemw + (size_t)(1 * CCH + c0 + laneM) * WPITCH + klo;
        const float* s3 = smemw + (size_t)(2 * CCH + c0 + laneM) * WPITCH + klo;
        const float* s4 = smemw + (size_t)(3 * CCH + c0 + laneM) * WPITCH + klo;
        #pragma unroll 8
        for (int k0 = 0; k0 < CCH; k0 += 4) {
            v2f A  = *(const v2f*)(xrow + k0 + klo);
            v2f B1 = *(const v2f*)(s1 + k0);
            v2f B2 = *(const v2f*)(s2 + k0);
            v2f B3 = *(const v2f*)(s3 + k0);
            v2f B4 = *(const v2f*)(s4 + k0);
            acc1 = __builtin_amdgcn_wmma_f32_16x16x4_f32(false, A, false, B1, (short)0, acc1, false, false);
            acc2 = __builtin_amdgcn_wmma_f32_16x16x4_f32(false, A, false, B2, (short)0, acc2, false, false);
            acc3 = __builtin_amdgcn_wmma_f32_16x16x4_f32(false, A, false, B3, (short)0, acc3, false, false);
            acc4 = __builtin_amdgcn_wmma_f32_16x16x4_f32(false, A, false, B4, (short)0, acc4, false, false);
        }
        const float b1v = bl1[c0 + laneM];
        const float b2v = bl2[c0 + laneM];
        const float b3v = br1[c0 + laneM];
        const float b4v = br2[c0 + laneM];
        v8bf lv, rv;
        #pragma unroll
        for (int v = 0; v < 8; ++v) {
            float lval = sigmoidf(acc1[v] + b1v) * (acc2[v] + b2v);
            float rval = sigmoidf(acc3[v] + b3v) * (acc4[v] + b4v);
            lv[v] = (__bf16)lval;
            rv[v] = (__bf16)rval;
        }
        // lane's channel = c0+laneM; its 8 accumulators are 8 consecutive cols
        const size_t off = (size_t)(c0 + laneM) * LL + (size_t)a * LDIM + b0 + mhi;
        *(v8bf*)(Lbf + off) = lv;   // 16B contiguous store per lane
        *(v8bf*)(Rbf + off) = rv;
    }
}

// ---------------------------------------------------------------------------
// Kernel 2: mega-fused triangle contraction + LayerNorm + output gating.
// One 256-thread block owns one 16x16 (i,j) tile across ALL 128 channels:
//   phase 1: each wave computes 16 channels of tri via bf16 WMMA (K=512),
//            double-buffered fragment loads, results into 256xWPITCH LDS tile.
//   phase 2: per-position LayerNorm over channels, in LDS.
//   phase 3: sigmoid(LN@Wg^T+bg) * (LN@Wo^T+bo) via f32 WMMA reading from LDS.
// tri never touches HBM; left/right panels (128MB bf16) live in the 192MB L2.
// ---------------------------------------------------------------------------
__global__ __launch_bounds__(256) void tri_kernel(
    const __bf16* __restrict__ Lbf, const __bf16* __restrict__ Rbf,
    const float* __restrict__ Wg, const float* __restrict__ bg,
    const float* __restrict__ Wo, const float* __restrict__ bo,
    const float* __restrict__ lng, const float* __restrict__ lnb,
    float* __restrict__ out)
{
    extern __shared__ float smem[];          // 256 * WPITCH floats
    const int lane  = threadIdx.x & 31;
    const int wave  = threadIdx.x >> 5;
    const int laneM = lane & 15;
    const int s8    = (lane < 16) ? 0 : 8;   // bf16 A frag hi half-wave: K+8
    const int s16   = (lane < 16) ? 0 : 16;  // unused; B frag uses s8 (K+8) like A
    const int mhi   = (lane < 16) ? 0 : 8;

    const int i0 = (blockIdx.x >> 5) * 16;
    const int j0 = (blockIdx.x & 31) * 16;

    // ---- phase 1: tri[i0:+16, j0:+16, c] for this wave's 16 channels ----
    for (int cc = 0; cc < 16; ++cc) {
        const int ch = wave * 16 + cc;
        const __bf16* pA = Lbf + (size_t)ch * LL + (size_t)(i0 + laneM) * LDIM;
        const __bf16* pB = Rbf + (size_t)ch * LL + (size_t)(j0 + laneM) * LDIM;
        if (cc + 1 < 16) {                   // global_prefetch_b8 next panel
            __builtin_prefetch(pA + LL, 0, 0);
            __builtin_prefetch(pB + LL, 0, 0);
        }
        // software-pipelined K loop: next fragments in flight during WMMA
        v8bf a0 = *(const v8bf*)(pA + s8);
        v8bf a1 = *(const v8bf*)(pA + 16 + s8);
        v8bf b0 = *(const v8bf*)(pB + s8);
        v8bf b1 = *(const v8bf*)(pB + 16 + s8);
        v8f acc = {};
        #pragma unroll 4
        for (int k0 = 0; k0 < LDIM; k0 += 32) {
            const int kn = (k0 + 32 < LDIM) ? (k0 + 32) : k0;   // clamp (no OOB)
            v8bf na0 = *(const v8bf*)(pA + kn + s8);
            v8bf na1 = *(const v8bf*)(pA + kn + 16 + s8);
            v8bf nb0 = *(const v8bf*)(pB + kn + s8);
            v8bf nb1 = *(const v8bf*)(pB + kn + 16 + s8);
            v16bf A = __builtin_shufflevector(a0, a1, 0,1,2,3,4,5,6,7,8,9,10,11,12,13,14,15);
            v16bf B = __builtin_shufflevector(b0, b1, 0,1,2,3,4,5,6,7,8,9,10,11,12,13,14,15);
            acc = __builtin_amdgcn_wmma_f32_16x16x32_bf16(false, A, false, B, (short)0, acc, false, false);
            a0 = na0; a1 = na1; b0 = nb0; b1 = nb1;
        }
        #pragma unroll
        for (int v = 0; v < 8; ++v) {
            const int pos = (mhi + v) * 16 + laneM;   // i_local*16 + j_local
            smem[pos * WPITCH + ch] = acc[v];
        }
    }
    __syncthreads();

    // ---- phase 2: LayerNorm over channels, one thread per position ----
    {
        float* row = smem + (size_t)threadIdx.x * WPITCH;
        float mu = 0.f;
        #pragma unroll 8
        for (int c = 0; c < CCH; ++c) mu += row[c];
        mu *= (1.0f / CCH);
        float var = 0.f;
        #pragma unroll 8
        for (int c = 0; c < CCH; ++c) { float d = row[c] - mu; var += d * d; }
        var *= (1.0f / CCH);
        const float rs = rsqrtf(var + LN_EPS);
        #pragma unroll 8
        for (int c = 0; c < CCH; ++c)
            row[c] = (row[c] - mu) * rs * lng[c] + lnb[c];
    }
    __syncthreads();

    // ---- phase 3: output gating, f32 WMMA A-fragments from LDS ----
    const int klo = (lane < 16) ? 0 : 2;
    for (int t = 0; t < 2; ++t) {
        const int mt = wave * 2 + t;             // LDS rows mt*16..+15 (i_local = mt)
        const float* arow = smem + (size_t)(mt * 16 + laneM) * WPITCH;
        for (int nt = 0; nt < 8; ++nt) {
            const int c0 = nt * 16;
            v8f accG = {}, accO = {};
            const float* wg = Wg + (size_t)(c0 + laneM) * CCH;
            const float* wo = Wo + (size_t)(c0 + laneM) * CCH;
            #pragma unroll 8
            for (int k0 = 0; k0 < CCH; k0 += 4) {
                v2f A  = *(const v2f*)(arow + k0 + klo);
                v2f Bg = *(const v2f*)(wg + k0 + klo);
                v2f Bo = *(const v2f*)(wo + k0 + klo);
                accG = __builtin_amdgcn_wmma_f32_16x16x4_f32(false, A, false, Bg, (short)0, accG, false, false);
                accO = __builtin_amdgcn_wmma_f32_16x16x4_f32(false, A, false, Bo, (short)0, accO, false, false);
            }
            const float bgv = bg[c0 + laneM];
            const float bov = bo[c0 + laneM];
            #pragma unroll
            for (int v = 0; v < 8; ++v) {
                const int jl = mhi + v;          // j_local
                const float val = sigmoidf(accG[v] + bgv) * (accO[v] + bov);
                out[(size_t)((i0 + mt) * LDIM + j0 + jl) * CCH + c0 + laneM] = val;
            }
        }
    }
}

// ---------------------------------------------------------------------------
extern "C" void kernel_launch(void* const* d_in, const int* in_sizes, int n_in,
                              void* d_out, int out_size, void* d_ws, size_t ws_size,
                              hipStream_t stream) {
    (void)in_sizes; (void)n_in; (void)out_size; (void)ws_size;
    const float* x   = (const float*)d_in[0];
    const float* Wl1 = (const float*)d_in[1];
    const float* bl1 = (const float*)d_in[2];
    const float* Wl2 = (const float*)d_in[3];
    const float* bl2 = (const float*)d_in[4];
    const float* Wr1 = (const float*)d_in[5];
    const float* br1 = (const float*)d_in[6];
    const float* Wr2 = (const float*)d_in[7];
    const float* br2 = (const float*)d_in[8];
    const float* Wg  = (const float*)d_in[9];
    const float* bg  = (const float*)d_in[10];
    const float* Wo  = (const float*)d_in[11];
    const float* bo  = (const float*)d_in[12];
    const float* lng = (const float*)d_in[13];
    const float* lnb = (const float*)d_in[14];

    // workspace: left/right in bf16, [c][row][k] layout, 64MB each
    __bf16* Lbf = (__bf16*)d_ws;
    __bf16* Rbf = (__bf16*)((char*)d_ws + (size_t)CCH * LL * 2);

    // kernel 1: 16384 position-tiles, 8 waves/block; 4*128*132*4 B LDS for weights
    proj_kernel<<<2048, 256, 4 * CCH * WPITCH * sizeof(float), stream>>>(
        x, Wl1, bl1, Wl2, bl2, Wr1, br1, Wr2, br2, Lbf, Rbf);

    // kernel 2: one block per 16x16 (i,j) tile; 256*132*4 = 135168 B LDS
    tri_kernel<<<1024, 256, 256 * WPITCH * sizeof(float), stream>>>(
        Lbf, Rbf, Wg, bg, Wo, bo, lng, lnb, (float*)d_out);
}
